// CrossAttention_concat_10754598109600
// MI455X (gfx1250) — matrix operand order repaired and checked by reading, not compiled, not run on hardware
//
#include <hip/hip_runtime.h>

// Problem constants (from reference)
#define BB   8
#define NN   2048
#define MM   2048
#define DIN  256
#define EE   128
#define ALPHA 0.01f

typedef __attribute__((ext_vector_type(16))) _Float16 v16h;
typedef __attribute__((ext_vector_type(4)))  _Float16 v4h;
typedef __attribute__((ext_vector_type(8)))  float    v8f;

// ---------------------------------------------------------------------------
// Fragment-layout helpers (ISA 7.12.2, wave32 16-bit operands)
//   B 32x16 (KxN): lane = (k&16) | (n&15), j = k&15.
//   A 16x32 (MxK): lane = (k&16) | m, j = k&15.
//   C/D 16x16 f32: lanes 0-15: VGPR r -> M=r; lanes 16-31: M=8+r; N=lane&15.
// For a 4-aligned km group g (km=4g): lane-half = g>>2, j0 = (g&3)*4
// -> one float4 global load becomes one 8-byte LDS store.
// ---------------------------------------------------------------------------

__device__ __forceinline__ v4h cvt4(float4 f) {
    v4h h;
    h.x = (_Float16)f.x; h.y = (_Float16)f.y;
    h.z = (_Float16)f.z; h.w = (_Float16)f.w;
    return h;
}

__device__ __forceinline__ v8f wmma16(v16h a, v16h b, v8f c) {
    return __builtin_amdgcn_wmma_f32_16x16x32_f16(false, a, false, b, (short)0, c,
                                                  false, false);
}

// ---------------------------------------------------------------------------
// Kernel 1: W1 (256x128 f32) -> B-fragment-swizzled f16 W1f[kt(8)][ec(8)][32][16]
// and vtop = W1 @ a1[:128]
// ---------------------------------------------------------------------------
__global__ __launch_bounds__(256) void prep_kernel(const float* __restrict__ W1,
                                                   const float* __restrict__ a1,
                                                   _Float16* __restrict__ W1f,
                                                   float* __restrict__ vtop) {
    int idx = blockIdx.x * 256 + threadIdx.x;           // 32768 elements
    if (idx < DIN * EE) {
        int k = idx / EE, e = idx % EE;
        int kt = k >> 5, ec = e >> 4;
        int l  = (k & 16) | (e & 15);
        int j  = k & 15;
        W1f[(((size_t)(kt * 8 + ec)) * 32 + l) * 16 + j] = (_Float16)W1[idx];
    }
    if (blockIdx.x == 0) {
        int d = threadIdx.x;                            // 256 threads == DIN
        float s = 0.f;
        #pragma unroll 4
        for (int e = 0; e < EE; ++e) s += W1[d * EE + e] * a1[e];
        vtop[d] = s;
    }
}

// ---------------------------------------------------------------------------
// Kernel 2: att1[row] = dot(in1[row,:256], vtop)   (wave per row, wave32)
// ---------------------------------------------------------------------------
__global__ __launch_bounds__(256) void att1_kernel(const float* __restrict__ in1,
                                                   const float* __restrict__ vtop,
                                                   float* __restrict__ att1) {
    int wave = threadIdx.x >> 5, lane = threadIdx.x & 31;
    int row  = blockIdx.x * 8 + wave;                   // row in [0, B*N)
    const float* p = in1 + (size_t)row * DIN;
    float s = 0.f;
    #pragma unroll
    for (int k = lane; k < DIN; k += 32) s += p[k] * vtop[k];
    #pragma unroll
    for (int off = 16; off > 0; off >>= 1) s += __shfl_xor(s, off, 32);
    if (lane == 0) att1[row] = s;
}

// ---------------------------------------------------------------------------
// Kernel 3: WMMA projection (double-buffered).  Wh2 = in2 @ W1 written as
// B-fragment-swizzled f16 Wh2f[b][kt2(64)][ec(8)][32][16]; att2 fused in.
// block = 256 threads = 8 waves; 16 rows x 128 cols; wave w owns ec=w.
// ---------------------------------------------------------------------------
__global__ __launch_bounds__(256) void proj_wmma_kernel(const float* __restrict__ in2,
                                                        const _Float16* __restrict__ W1f,
                                                        const float* __restrict__ a1,
                                                        _Float16* __restrict__ Wh2f,
                                                        float* __restrict__ att2) {
    __shared__ __align__(32) _Float16 Abuf[2][32][16];  // double-buffered A frag (2 KB)
    __shared__ float satt2[16];
    const int tid  = threadIdx.x;
    const int wave = tid >> 5, lane = tid & 31;
    const int row0 = blockIdx.x * 16;                   // flattened b*M + m
    const int b    = row0 / MM;
    const int mm0  = row0 % MM;
    const int ncol = lane & 15, half = lane >> 4;
    // staging coords (threads 0..127 stage, one float4 each)
    const int srr = tid >> 3, sg = tid & 7;
    const int sl  = (sg >> 2) * 16 + srr;
    const int sj  = (sg & 3) * 4;
    const float* srcRow = in2 + (size_t)(row0 + srr) * DIN + sg * 4;

    if (tid < 16) satt2[tid] = 0.f;

    auto stage = [&](int kt, _Float16 (*buf)[16]) {
        if (tid < 128) {
            float4 v = *(const float4*)(srcRow + kt * 32);
            *(v4h*)&buf[sl][sj] = cvt4(v);
        }
    };
    auto compute = [&](int kt, _Float16 (*buf)[16], v8f& c) {
        v16h a  = *(const v16h*)buf[lane];              // 2x ds_load_b128
        v16h bf = *(const v16h*)(W1f + (((size_t)(kt * 8 + wave)) * 32 + lane) * 16);
        c = wmma16(a, bf, c);
    };

    v8f c = {};
    stage(0, Abuf[0]);
    __syncthreads();
    for (int kt = 0; kt < DIN / 32; kt += 2) {          // 8 tiles, 2 per round
        stage(kt + 1, Abuf[1]);                         // prefetch next into other buf
        compute(kt, Abuf[0], c);
        __syncthreads();
        if (kt + 2 < DIN / 32) stage(kt + 2, Abuf[0]);
        compute(kt + 1, Abuf[1], c);
        __syncthreads();
    }

    // --- store Wh2 tile into B-fragment-swizzled layout (f16) ---
    #pragma unroll
    for (int r = 0; r < 8; ++r) {
        int mm  = mm0 + half * 8 + r;                   // row within batch
        int kt2 = mm >> 5;
        int l   = (mm & 16) | ncol;
        int j   = mm & 15;
        Wh2f[((((size_t)b * 64 + kt2) * 8 + wave) * 32 + l) * 16 + j] = (_Float16)c[r];
    }

    // --- att2 reduction: att2[m] = sum_e Wh2[m][e] * a_bot[e] ---
    const float ab = a1[EE + wave * 16 + ncol];
    #pragma unroll
    for (int r = 0; r < 8; ++r) {
        float p = c[r] * ab;
        #pragma unroll
        for (int off = 8; off > 0; off >>= 1) p += __shfl_xor(p, off, 32);
        if (ncol == 0) atomicAdd(&satt2[half * 8 + r], p);
    }
    __syncthreads();
    if (tid < 16) att2[row0 + tid] = satt2[tid];
}

// ---------------------------------------------------------------------------
// Kernel 4: row softmax (float4-vectorized).  One block per (b,n) row.
// ---------------------------------------------------------------------------
__global__ __launch_bounds__(256) void softmax_kernel(const float* __restrict__ att1,
                                                      const float* __restrict__ att2,
                                                      const float* __restrict__ mask,
                                                      float* __restrict__ probs) {
    __shared__ float4 sc[MM / 4];                       // 8 KB
    __shared__ float red[256];
    const int tid = threadIdx.x;
    const int b   = blockIdx.x / NN;
    const float a1v = att1[blockIdx.x];
    const float4* at2 = (const float4*)(att2 + (size_t)b * MM);
    const float4* mk  = (const float4*)(mask + (size_t)b * MM);

    float lmax = -1e30f;
    #pragma unroll
    for (int m4 = tid; m4 < MM / 4; m4 += 256) {        // 2 iterations
        float4 t = at2[m4], q = mk[m4], e;
        e.x = a1v + t.x; e.x = (e.x > 0.f ? e.x : ALPHA * e.x) + q.x;
        e.y = a1v + t.y; e.y = (e.y > 0.f ? e.y : ALPHA * e.y) + q.y;
        e.z = a1v + t.z; e.z = (e.z > 0.f ? e.z : ALPHA * e.z) + q.z;
        e.w = a1v + t.w; e.w = (e.w > 0.f ? e.w : ALPHA * e.w) + q.w;
        sc[m4] = e;
        lmax = fmaxf(lmax, fmaxf(fmaxf(e.x, e.y), fmaxf(e.z, e.w)));
    }
    red[tid] = lmax; __syncthreads();
    #pragma unroll
    for (int st = 128; st > 0; st >>= 1) {
        if (tid < st) red[tid] = fmaxf(red[tid], red[tid + st]);
        __syncthreads();
    }
    const float rmax = red[0]; __syncthreads();

    float lsum = 0.f;
    #pragma unroll
    for (int m4 = tid; m4 < MM / 4; m4 += 256) {
        float4 e = sc[m4], p;
        p.x = __expf(e.x - rmax); p.y = __expf(e.y - rmax);
        p.z = __expf(e.z - rmax); p.w = __expf(e.w - rmax);
        sc[m4] = p;
        lsum += (p.x + p.y) + (p.z + p.w);
    }
    red[tid] = lsum; __syncthreads();
    #pragma unroll
    for (int st = 128; st > 0; st >>= 1) {
        if (tid < st) red[tid] += red[tid + st];
        __syncthreads();
    }
    const float inv = 1.0f / red[0];

    float4* out = (float4*)(probs + (size_t)blockIdx.x * MM);
    #pragma unroll
    for (int m4 = tid; m4 < MM / 4; m4 += 256) {
        float4 p = sc[m4];
        p.x *= inv; p.y *= inv; p.z *= inv; p.w *= inv;
        out[m4] = p;                                    // global_store_b128
    }
}

// ---------------------------------------------------------------------------
// Kernel 5: WMMA AV GEMM (double-buffered).
// context[32 x 128] = probs[32 x 2048] @ Wh2[2048 x 128].
// 32 rows/block (2 row-groups), K=64 per tile -> 4 WMMAs per barrier, each B
// fragment reused by 2 WMMAs.  Next tile's probs loads are issued before this
// tile's WMMAs so HBM latency hides behind the MMA pipe.
// ---------------------------------------------------------------------------
__global__ __launch_bounds__(256) void context_wmma_kernel(const float* __restrict__ probs,
                                                           const _Float16* __restrict__ Wh2f,
                                                           float* __restrict__ ctx) {
    __shared__ __align__(32) _Float16 Abuf[2][2][2][32][16]; // [buf][rg][kb] 8 KB
    const int tid  = threadIdx.x;
    const int wave = tid >> 5, lane = tid & 31;
    const int row0 = blockIdx.x * 32;                   // flattened b*N + n
    const int b    = row0 / NN;                         // 32 | N -> uniform
    const int ncol = lane & 15, half = lane >> 4;
    const _Float16* Wb = Wh2f + (size_t)b * MM * EE;    // batch's fragment array

    // staging coords: thread t in {0,1}: i4 = tid + t*256 over 512 float4
    auto stage = [&](int kt2, _Float16 (*buf)[2][32][16]) {
        const int k0 = kt2 * 64;
        #pragma unroll
        for (int t = 0; t < 2; ++t) {
            int i4 = tid + t * 256;
            int rg = i4 >> 8, kb = (i4 >> 7) & 1;
            int rr = (i4 >> 3) & 15, g = i4 & 7;
            float4 v = *(const float4*)(probs +
                (size_t)(row0 + rg * 16 + rr) * MM + k0 + kb * 32 + g * 4);
            int l = (g >> 2) * 16 + rr;
            int j = (g & 3) * 4;
            *(v4h*)&buf[rg][kb][l][j] = cvt4(v);
        }
    };
    auto compute = [&](int kt2, _Float16 (*buf)[2][32][16], v8f& c0, v8f& c1) {
        #pragma unroll
        for (int kb = 0; kb < 2; ++kb) {
            v16h bf = *(const v16h*)(Wb +
                (((size_t)(kt2 * 2 + kb) * 8 + wave) * 32 + lane) * 16);
            v16h a0 = *(const v16h*)buf[0][kb][lane];
            c0 = wmma16(a0, bf, c0);
            v16h a1 = *(const v16h*)buf[1][kb][lane];
            c1 = wmma16(a1, bf, c1);
        }
    };

    v8f c0 = {}, c1 = {};
    stage(0, Abuf[0]);
    __syncthreads();
    for (int kt2 = 0; kt2 < MM / 64; kt2 += 2) {        // 32 tiles, 2 per round
        stage(kt2 + 1, Abuf[1]);                        // prefetch (kt2+1 <= 31 always)
        compute(kt2, Abuf[0], c0, c1);
        __syncthreads();
        if (kt2 + 2 < MM / 64) stage(kt2 + 2, Abuf[0]);
        compute(kt2 + 1, Abuf[1], c0, c1);
        __syncthreads();
    }
    #pragma unroll
    for (int r = 0; r < 8; ++r) {
        ctx[(size_t)(row0 + half * 8 + r) * EE + wave * 16 + ncol]      = c0[r];
        ctx[(size_t)(row0 + 16 + half * 8 + r) * EE + wave * 16 + ncol] = c1[r];
    }
}

// ---------------------------------------------------------------------------
extern "C" void kernel_launch(void* const* d_in, const int* in_sizes, int n_in,
                              void* d_out, int out_size, void* d_ws, size_t ws_size,
                              hipStream_t stream) {
    const float* in1  = (const float*)d_in[0];   // (B,N,256)
    const float* in2  = (const float*)d_in[1];   // (B,M,256)
    const float* mask = (const float*)d_in[2];   // (B,1,M)
    const float* W1   = (const float*)d_in[3];   // (256,128)
    const float* a1   = (const float*)d_in[4];   // (256,1)

    // d_out: context (B*N*128 f32) then probs (B*N*M f32)
    float* ctx   = (float*)d_out;
    float* probs = ctx + (size_t)BB * NN * EE;

    // Workspace carve-up (~4.3 MB), all 32B-aligned chunks
    char* ws = (char*)d_ws;
    _Float16* W1f  = (_Float16*)ws;              ws += (size_t)DIN * EE * sizeof(_Float16);
    float*    vtop = (float*)ws;                 ws += (size_t)DIN * sizeof(float);
    float*    att1 = (float*)ws;                 ws += (size_t)BB * NN * sizeof(float);
    float*    att2 = (float*)ws;                 ws += (size_t)BB * MM * sizeof(float);
    _Float16* Wh2f = (_Float16*)ws;              // B*M*E f16 fragments (4 MB)

    // 1. W1 -> fragment-swizzled f16; vtop = W1 @ a_top
    prep_kernel<<<dim3((DIN * EE) / 256), dim3(256), 0, stream>>>(W1, a1, W1f, vtop);
    // 2. att1 = in1 @ vtop
    att1_kernel<<<dim3(BB * NN / 8), dim3(256), 0, stream>>>(in1, vtop, att1);
    // 3. Wh2f = frag(f16(in2 @ W1)); att2 fused          [WMMA]
    proj_wmma_kernel<<<dim3(BB * MM / 16), dim3(256), 0, stream>>>(in2, W1f, a1, Wh2f, att2);
    // 4. probs = softmax(leakyrelu(att1 + att2^T) + mask)
    softmax_kernel<<<dim3(BB * NN), dim3(256), 0, stream>>>(att1, att2, mask, probs);
    // 5. context = probs @ Wh2                            [WMMA]
    context_wmma_kernel<<<dim3(BB * NN / 32), dim3(256), 0, stream>>>(probs, Wh2f, ctx);
}
